// OT_Loss_67834713473696
// MI455X (gfx1250) — matrix operand order repaired and checked by reading, not compiled, not run on hardware
//
#include <hip/hip_runtime.h>
#include <hip/hip_bf16.h>
#include <math.h>

// ---------------------------------------------------------------------------
// Sinkhorn OT loss, N=8192, reg=10, 100 iterations.
// Memory-bound: 100 iters x (K + K^T) in f16 = 25.6 GB => ~1.1 ms @ 23.3 TB/s.
// Matvecs use v_wmma_f32_16x16x32_f16 with the vector replicated across the
// 16 B-columns (all D columns equal the matvec result), f32 accumulation.
// ---------------------------------------------------------------------------

#define NN 8192
#define REGP 10.0f
#define NITER 100
#define MEPS 1e-16f
#define VSCALE 4096.0f   // power-of-2 scaling for f16 u/v copies (exact)

typedef __attribute__((ext_vector_type(16))) _Float16 v16h;
typedef __attribute__((ext_vector_type(8)))  _Float16 v8h;
typedef __attribute__((ext_vector_type(8)))  float    v8f;

// ---------------- block reduction helpers ----------------
template<int NT>
__device__ inline float blkSum(float v, float* sd) {
  int t = threadIdx.x;
  sd[t] = v; __syncthreads();
  #pragma unroll
  for (int s = NT / 2; s > 0; s >>= 1) {
    if (t < s) sd[t] += sd[t + s];
    __syncthreads();
  }
  float r = sd[0]; __syncthreads();
  return r;
}
template<int NT>
__device__ inline float blkMax(float v, float* sd) {
  int t = threadIdx.x;
  sd[t] = v; __syncthreads();
  #pragma unroll
  for (int s = NT / 2; s > 0; s >>= 1) {
    if (t < s) sd[t] = fmaxf(sd[t], sd[t + s]);
    __syncthreads();
  }
  float r = sd[0]; __syncthreads();
  return r;
}
template<int NT>
__device__ inline float blkMin(float v, float* sd) {
  int t = threadIdx.x;
  sd[t] = v; __syncthreads();
  #pragma unroll
  for (int s = NT / 2; s > 0; s >>= 1) {
    if (t < s) sd[t] = fminf(sd[t], sd[t + s]);
    __syncthreads();
  }
  float r = sd[0]; __syncthreads();
  return r;
}

// ---------------- K0: score statistics (single block) ----------------
// scal[0]=tmax [1]=t_sumexp [2]=smax [3]=s_sumexp [4]=tmin [5]=smin
// scal[6]=sum(ss) [7]=dist_max (float bits, zeroed here) [8]=score_cost.max()
__global__ __launch_bounds__(1024)
void k_stats(const float* __restrict__ ts, const float* __restrict__ ss,
             float* __restrict__ scal) {
  __shared__ float sd[1024];
  int t = threadIdx.x;
  float tmax = -1e30f, tmin = 1e30f, smax = -1e30f, smin = 1e30f, ssum = 0.f;
  for (int i = t; i < NN; i += 1024) {
    float a = ts[i], b = ss[i];
    tmax = fmaxf(tmax, a); tmin = fminf(tmin, a);
    smax = fmaxf(smax, b); smin = fminf(smin, b);
    ssum += b;
  }
  tmax = blkMax<1024>(tmax, sd);
  tmin = blkMin<1024>(tmin, sd);
  smax = blkMax<1024>(smax, sd);
  smin = blkMin<1024>(smin, sd);
  ssum = blkSum<1024>(ssum, sd);

  float tse = 0.f, sse = 0.f;
  for (int i = t; i < NN; i += 1024) {
    tse += __expf(ts[i] - tmax);
    sse += __expf(ss[i] - smax);
  }
  tse = blkSum<1024>(tse, sd);
  sse = blkSum<1024>(sse, sd);

  if (t == 0) {
    scal[0] = tmax; scal[1] = tse; scal[2] = smax; scal[3] = sse;
    scal[4] = tmin; scal[5] = smin; scal[6] = ssum;
    ((unsigned*)scal)[7] = 0u;                    // init dist-max atomic slot
    float d1 = tmax - smin, d2 = smax - tmin;     // closed-form score_cost.max()
    scal[8] = fmaxf(d1 * d1, d2 * d2);
  }
}

// ---------------- K1: max pairwise squared distance (brute force) -----------
__global__ __launch_bounds__(256)
void k_distmax(const float* __restrict__ pts, float* __restrict__ scal) {
  __shared__ float sd[256];
  int i = blockIdx.x * 256 + threadIdx.x;     // gridDim.x = 32  -> i in [0,8192)
  int j0 = blockIdx.y * 1024;                 // gridDim.y = 8
  float xi = pts[2 * i], yi = pts[2 * i + 1];
  float m = 0.f;
  for (int j = j0; j < j0 + 1024; ++j) {      // broadcast loads within wave
    float dx = pts[2 * j] - xi;
    float dy = pts[2 * j + 1] - yi;
    m = fmaxf(m, dx * dx + dy * dy);
  }
  m = blkMax<256>(m, sd);
  if (threadIdx.x == 0)
    atomicMax((unsigned*)(scal + 7), __float_as_uint(m)); // valid: m >= 0
}

// ---------------- K2: softmax probs a,b + u0 init ----------------
__global__ __launch_bounds__(256)
void k_init(const float* __restrict__ ts, const float* __restrict__ ss,
            const float* __restrict__ scal,
            float* __restrict__ a, float* __restrict__ b,
            float* __restrict__ u, _Float16* __restrict__ u16) {
  int i = blockIdx.x * 256 + threadIdx.x;
  float tmax = scal[0], tse = scal[1], smax = scal[2], sse = scal[3];
  a[i] = __expf(ts[i] - tmax) / tse;   // t_prob (row marginal)
  b[i] = __expf(ss[i] - smax) / sse;   // s_prob (col marginal)
  u[i] = 1.0f / (float)NN;
  u16[i] = (_Float16)(VSCALE / (float)NN);
}

// ---------------- K3: materialize K (or K^T) in f16, row-major --------------
// dst[r*NN+c] = exp(-( dist(r,c)/dmax + (rs[r]-cs[c])^2/smax ) / reg)
// K : rs = t_scores, cs = s_scores.   K^T : rs = s_scores, cs = t_scores.
__global__ __launch_bounds__(256)
void k_gen(_Float16* __restrict__ dst,
           const float* __restrict__ rs, const float* __restrict__ cs,
           const float* __restrict__ pts, const float* __restrict__ scal) {
  size_t gid  = (size_t)blockIdx.x * 256 + threadIdx.x;
  size_t base = gid * 8;                  // 8 contiguous f16 per thread (16 B)
  int r  = (int)(base >> 13);             // / 8192
  int c0 = (int)(base & (NN - 1));
  float invD = 1.0f / scal[7];
  float invS = 1.0f / scal[8];
  float rv = rs[r];
  float rx = pts[2 * r], ry = pts[2 * r + 1];
  v8h out;
  #pragma unroll
  for (int e = 0; e < 8; ++e) {
    int c = c0 + e;
    float dx = pts[2 * c] - rx;
    float dy = pts[2 * c + 1] - ry;
    float sdf = rv - cs[c];
    float cost = (dx * dx + dy * dy) * invD + sdf * sdf * invS;
    out[e] = (_Float16)__expf(cost * (-1.0f / REGP));
  }
  *(v8h*)(dst + base) = out;
}

// ---------------- K4: WMMA matvec + fused Sinkhorn update -------------------
// out = tgt / (M @ x + eps).  M row-major NN x NN f16; x16 = x * VSCALE (f16).
// One block per 16-row strip; 8 waves split k; A = M-tile, B = x replicated
// across 16 columns so every column of D is the matvec result.
__global__ __launch_bounds__(256)
void k_matvec(const _Float16* __restrict__ M, const _Float16* __restrict__ x16,
              const float* __restrict__ tgt,
              float* __restrict__ outF, _Float16* __restrict__ out16) {
  __shared__ float red[8][16];
  const int lane = threadIdx.x & 31;
  const int wave = threadIdx.x >> 5;
  const int half = lane >> 4;
  const int r0   = blockIdx.x * 16;
  const int row  = r0 + (lane & 15);

  // Dense 16-bit A 16x32 layout (ISA 7.12.2): lanes 0-15 hold K {0..7,16..23},
  // lanes 16-31 hold K {8..15,24..31}.
  const int aOff0 = 8 * half;
  const int aOff1 = 16 + 8 * half;
  // Dense 16-bit B 32x16 (same K order as A): lanes 0-15 K {0..7,16..23},
  // lanes 16-31 K {8..15,24..31}; replicated vector => identical per half-wave.
  const int bOff = 8 * half;

  const _Float16* Arow = M + (size_t)row * NN;

  v8f c = {};
  const int kb = wave * (NN / 8);      // 1024-wide k-chunk per wave
  #pragma unroll 2
  for (int k0 = kb; k0 < kb + NN / 8; k0 += 32) {
    v8h alo = *(const v8h*)(Arow + k0 + aOff0);   // global_load_b128
    v8h ahi = *(const v8h*)(Arow + k0 + aOff1);   // global_load_b128
    v8h blo = *(const v8h*)(x16 + k0 + bOff);     // broadcast within half-wave
    v8h bhi = *(const v8h*)(x16 + k0 + bOff + 16);
    v16h A = __builtin_shufflevector(alo, ahi, 0,1,2,3,4,5,6,7,8,9,10,11,12,13,14,15);
    v16h B = __builtin_shufflevector(blo, bhi, 0,1,2,3,4,5,6,7,8,9,10,11,12,13,14,15);
    c = __builtin_amdgcn_wmma_f32_16x16x32_f16(false, A, false, B,
                                               (short)0, c, false, false);
  }

  // D layout: lane 0 VGPR r = row r (col 0), lane 16 VGPR r = row 8+r.
  if ((lane & 15) == 0) {
    #pragma unroll
    for (int r = 0; r < 8; ++r) red[wave][half * 8 + r] = c[r];
  }
  __syncthreads();
  if (threadIdx.x < 16) {
    float s = 0.f;
    #pragma unroll
    for (int w = 0; w < 8; ++w) s += red[w][threadIdx.x];
    int orow = r0 + threadIdx.x;
    float dot = s * (1.0f / VSCALE);           // undo exact power-of-2 scale
    float val = tgt[orow] / (dot + MEPS);
    outF[orow]  = val;
    out16[orow] = (_Float16)(val * VSCALE);
  }
}

// ---------------- K5: beta = reg*log(v+eps), loss contraction ---------------
__global__ __launch_bounds__(1024)
void k_final(const float* __restrict__ ss, const float* __restrict__ v,
             const float* __restrict__ scal, float* __restrict__ out) {
  __shared__ float sd[1024];
  int t = threadIdx.x;
  float T = 0.f;
  for (int i = t; i < NN; i += 1024) {
    float beta = REGP * __logf(v[i] + MEPS);
    T += ss[i] * beta;
  }
  T = blkSum<1024>(T, sd);
  float sc = scal[6];
  float denom = sc * sc + 1e-8f;
  float loss = 0.f;
  for (int i = t; i < NN; i += 1024) {
    float beta = REGP * __logf(v[i] + MEPS);
    float ig = (sc / denom) * beta - T / denom;   // im_grad[i]
    loss += ss[i] * ig;
  }
  loss = blkSum<1024>(loss, sd);
  if (t == 0) out[0] = loss;
}

// ---------------------------------------------------------------------------
extern "C" void kernel_launch(void* const* d_in, const int* in_sizes, int n_in,
                              void* d_out, int out_size, void* d_ws, size_t ws_size,
                              hipStream_t stream) {
  const float* ts  = (const float*)d_in[0];   // t_scores (8192)
  const float* ss  = (const float*)d_in[1];   // s_scores (8192)
  const float* pts = (const float*)d_in[2];   // pts (8192,2) interleaved
  float* out = (float*)d_out;

  // Workspace layout (needs ~256 MB + 164 KB):
  char* w = (char*)d_ws;
  float* scal = (float*)w;                               // 256 B scalar slab
  float*    a   = (float*)(w + 256);                     // 32 KB
  float*    b   = a + NN;                                // 32 KB
  float*    u   = b + NN;                                // 32 KB
  float*    v   = u + NN;                                // 32 KB
  _Float16* u16 = (_Float16*)(v + NN);                   // 16 KB
  _Float16* v16 = u16 + NN;                              // 16 KB
  size_t off = 256 + 4 * (size_t)NN * 4 + 2 * (size_t)NN * 2;
  off = (off + 255) & ~(size_t)255;
  _Float16* K  = (_Float16*)(w + off);                   // 128 MB
  _Float16* KT = K + (size_t)NN * NN;                    // 128 MB

  k_stats<<<1, 1024, 0, stream>>>(ts, ss, scal);
  k_distmax<<<dim3(32, 8), 256, 0, stream>>>(pts, scal);
  k_init<<<NN / 256, 256, 0, stream>>>(ts, ss, scal, a, b, u, u16);

  const int genBlocks = (int)(((size_t)NN * NN / 8) / 256);  // 32768
  k_gen<<<genBlocks, 256, 0, stream>>>(K,  ts, ss, pts, scal);  // K[i][j]
  k_gen<<<genBlocks, 256, 0, stream>>>(KT, ss, ts, pts, scal);  // K^T[j][i]

  for (int it = 0; it < NITER; ++it) {
    // v = b / (u @ K + eps)  ==  b / (K^T u + eps)
    k_matvec<<<NN / 16, 256, 0, stream>>>(KT, u16, b, v, v16);
    // u = a / (K @ v + eps)
    k_matvec<<<NN / 16, 256, 0, stream>>>(K,  v16, a, u, u16);
  }

  k_final<<<1, 1024, 0, stream>>>(ss, v, scal, out);
}
